// DyConv2d_64063732187508
// MI455X (gfx1250) — matrix operand order repaired and checked by reading, not compiled, not run on hardware
//
#include <hip/hip_runtime.h>
#include <hip/hip_bf16.h>

// ---------- types ----------
typedef __attribute__((ext_vector_type(16))) __bf16        v16bf;
typedef __attribute__((ext_vector_type(8)))  float         v8f;
typedef __attribute__((ext_vector_type(4)))  unsigned int  u32x4;

union Frag16 {
    u32x4 q[2];
    v16bf v;
};

// round-to-nearest-even f32 -> bf16 (bits)
static __device__ __forceinline__ unsigned short f32_to_bf16(float f) {
    unsigned int u = __float_as_uint(f);
    u += 0x7FFFu + ((u >> 16) & 1u);
    return (unsigned short)(u >> 16);
}

// ---------- problem constants ----------
#define B_    16
#define CIN   128
#define COUT  128
#define KBR   4          // K branches
#define H_    64
#define W_    64
#define HW    (H_ * W_)
#define KTOT  1152       // 9 * 128, GEMM K
#define KSTEPS 36        // 1152 / 32
#define MT    8          // 128 / 16 M tiles
#define FRAG_ELEMS 512   // 32 lanes * 16 bf16 per A fragment
#define WA_ELEMS   ((size_t)B_ * MT * KSTEPS * FRAG_ELEMS)  // 2,359,296 bf16
#define LDS_CI 136       // padded ci stride (16B aligned reads, bank spread)

// =====================================================================
// K1: global average pool: pooled[b][c] = mean_{h,w} x[b][c][h][w]
// =====================================================================
__global__ void k_pool(const float* __restrict__ x, float* __restrict__ pooled) {
    const int bc = blockIdx.x;                 // 0 .. B*CIN-1
    const float* p = x + (size_t)bc * HW;
    float s = 0.f;
    for (int i = threadIdx.x; i < HW; i += 256) s += p[i];
    __shared__ float red[256];
    red[threadIdx.x] = s;
    __syncthreads();
    for (int off = 128; off > 0; off >>= 1) {
        if (threadIdx.x < (unsigned)off) red[threadIdx.x] += red[threadIdx.x + off];
        __syncthreads();
    }
    if (threadIdx.x == 0) pooled[bc] = red[0] * (1.0f / (float)HW);
}

// =====================================================================
// K2: SE MLP + softmax(T=30) + per-batch bias combine (one block)
// =====================================================================
__global__ void k_se(const float* __restrict__ pooled,
                     const float* __restrict__ w1,   // [33][128]
                     const float* __restrict__ w2,   // [4][33]
                     const float* __restrict__ b2,   // [4]
                     const float* __restrict__ bias, // [512]
                     float* __restrict__ attn,       // [16][4]  (ws)
                     float* __restrict__ bias_eff) { // [16][128](ws)
    __shared__ float hs[B_ * 33];
    __shared__ float at[B_ * KBR];
    const int tid = threadIdx.x;

    for (int t = tid; t < B_ * 33; t += 256) {
        const int b = t / 33, j = t % 33;
        float s = 0.f;
        for (int c = 0; c < CIN; ++c) s += pooled[b * CIN + c] * w1[j * CIN + c];
        hs[t] = s > 0.f ? s : 0.f;
    }
    __syncthreads();
    if (tid < B_ * KBR) {
        const int b = tid / KBR, k = tid % KBR;
        float s = b2[k];
        for (int j = 0; j < 33; ++j) s += hs[b * 33 + j] * w2[k * 33 + j];
        at[tid] = s * (1.0f / 30.0f);
    }
    __syncthreads();
    if (tid < B_) {
        const int b = tid;
        float m = at[b * KBR];
        for (int k = 1; k < KBR; ++k) m = fmaxf(m, at[b * KBR + k]);
        float e[KBR], sum = 0.f;
        for (int k = 0; k < KBR; ++k) { e[k] = __expf(at[b * KBR + k] - m); sum += e[k]; }
        const float inv = 1.0f / sum;
        for (int k = 0; k < KBR; ++k) {
            const float a = e[k] * inv;
            at[b * KBR + k] = a;
            attn[b * KBR + k] = a;
        }
    }
    __syncthreads();
    for (int t = tid; t < B_ * COUT; t += 256) {
        const int b = t / COUT, co = t % COUT;
        float s = 0.f;
        for (int k = 0; k < KBR; ++k) s += at[b * KBR + k] * bias[k * COUT + co];
        bias_eff[t] = s;
    }
}

// =====================================================================
// K3: combine weights over K branches with attn, convert to bf16, and
// swizzle into a contiguous-K WMMA A-fragment lane layout (k_conv's B uses the same).
//   K ordering: K = (r*3+s)*128 + ci  (ci innermost)
//   A layout:   lanes 0-15 (row M=lane)  hold K = {0..15}
//               lanes 16-31 (row M=lane-16) hold K = {16..31}
// Storage: wA[((b*8+m)*36+kk)*512 + lane*16 + e]  -> 32B/lane contiguous
// =====================================================================
__global__ void k_wcomb(const float* __restrict__ weight,  // [512][128][3][3]
                        const float* __restrict__ attn,    // [16][4]
                        unsigned short* __restrict__ wA) {
    const int kk = blockIdx.x;        // 0..35 K-step
    const int b  = blockIdx.y;        // 0..15 batch
    const int m    = threadIdx.x >> 5;   // 0..7  M tile (wave id)
    const int lane = threadIdx.x & 31;
    const int co   = m * 16 + (lane & 15);
    const int rs    = kk >> 2;           // filter tap 0..8
    const int cbase = (kk & 3) * 32;     // ci base within tap
    const int lo    = (lane & 16) ? 16 : 0;

    const float a0 = attn[b * KBR + 0];
    const float a1 = attn[b * KBR + 1];
    const float a2 = attn[b * KBR + 2];
    const float a3 = attn[b * KBR + 3];

    unsigned short* dst =
        wA + ((((size_t)b * MT + m) * KSTEPS + kk) * FRAG_ELEMS) + lane * 16;
    const size_t KSTRIDE = (size_t)COUT * CIN * 9;  // 147456 between branches

#pragma unroll
    for (int e = 0; e < 16; ++e) {
        const int kl = lo + e;   // K_local 0..31 per layout
        const int ci = cbase + kl;
        const size_t i0 = ((size_t)co * CIN + ci) * 9 + rs;
        const float v = a0 * weight[i0]
                      + a1 * weight[i0 + KSTRIDE]
                      + a2 * weight[i0 + 2 * KSTRIDE]
                      + a3 * weight[i0 + 3 * KSTRIDE];
        dst[e] = f32_to_bf16(v);
    }
}

// =====================================================================
// K4: implicit-GEMM conv row kernel. One block = (batch b, output row h).
// Block tile: M=128 (all out channels) x N=64 (full output row).
// 8 waves: wave = (mi2, ni): 4 M-groups x 2 N-groups; each wave owns a
// 32x32 C tile = 4 accumulators, 4 WMMA per K-step.
// The 36-step K loop is FULLY UNROLLED: all A addresses are immediate
// offsets off (wAb + lane*32); all B addresses are immediate DS offsets
// off a lane-dependent base -> no loop-carried address math, pinned
// accumulators, no register-rotation mov churn.
// =====================================================================
__global__ __launch_bounds__(256)
void k_conv(const float* __restrict__ x,
            const unsigned short* __restrict__ wA,
            const float* __restrict__ bias_eff,
            float* __restrict__ out) {
    __shared__ unsigned short xs[3 * 66 * LDS_CI];  // 53,856 B

    const int b = blockIdx.y;
    const int h = blockIdx.x;
    const int tid = threadIdx.x;
    const float* xb = x + (size_t)b * CIN * HW;

    // ---- stage 3 input rows (with halo + zero pad) as bf16, ci-contiguous ----
    for (int idx = tid; idx < 3 * 66 * CIN; idx += 256) {
        const int wcol = idx % 66;            // consecutive tid -> consecutive w (coalesced)
        const int t2   = idx / 66;
        const int r    = t2 % 3;
        const int ci   = t2 / 3;
        const int hin = h + r - 1;
        const int win = wcol - 1;
        float v = 0.f;
        if ((unsigned)hin < (unsigned)H_ && (unsigned)win < (unsigned)W_)
            v = xb[((size_t)ci * H_ + hin) * W_ + win];
        xs[(r * 66 + wcol) * LDS_CI + ci] = f32_to_bf16(v);
    }
    __syncthreads();

    const int wave = tid >> 5;
    const int lane = tid & 31;
    const int mi2  = wave >> 1;   // 0..3  -> M rows [mi2*32, +32)
    const int ni   = wave & 1;    // 0..1  -> N cols [ni*32, +32)
    const int lo16 = (lane & 16) ? 16 : 0;
    const int nc   = lane & 15;

    v8f acc[2][2];
#pragma unroll
    for (int mt = 0; mt < 2; ++mt)
#pragma unroll
        for (int nt = 0; nt < 2; ++nt) acc[mt][nt] = (v8f)0.0f;

    const unsigned short* wAb = wA + (size_t)b * MT * KSTEPS * FRAG_ELEMS;

    // loop-invariant fragment base pointers (all K-step addressing becomes
    // compile-time immediate offsets off these)
    const unsigned short* aBase[2];
#pragma unroll
    for (int mt = 0; mt < 2; ++mt)
        aBase[mt] = wAb + ((size_t)(mi2 * 2 + mt) * KSTEPS) * FRAG_ELEMS + lane * 16;

    const unsigned short* bBase[2];
#pragma unroll
    for (int nt = 0; nt < 2; ++nt)
        bBase[nt] = xs + (ni * 32 + nt * 16 + nc) * LDS_CI + lo16;

#pragma unroll
    for (int kk = 0; kk < KSTEPS; ++kk) {
        const int rs    = kk >> 2;          // constant per unrolled step
        const int cbase = (kk & 3) << 5;
        const int r = rs / 3;
        const int s = rs - r * 3;
        const int aOff = kk * FRAG_ELEMS;                  // immediate (bytes = kk*1024)
        const int bOff = (r * 66 + s) * LDS_CI + cbase;    // immediate DS offset

        Frag16 a[2], bf[2];
#pragma unroll
        for (int mt = 0; mt < 2; ++mt) {
            const u32x4* p = (const u32x4*)(aBase[mt] + aOff);
            a[mt].q[0] = p[0];
            a[mt].q[1] = p[1];
        }
        if ((kk & 3) == 0 && kk + 4 < KSTEPS) {  // warm L2 a few steps ahead
            __builtin_prefetch(aBase[0] + (kk + 4) * FRAG_ELEMS, 0, 1);
        }
#pragma unroll
        for (int nt = 0; nt < 2; ++nt) {
            const u32x4* p = (const u32x4*)(bBase[nt] + bOff);
            bf[nt].q[0] = p[0];
            bf[nt].q[1] = p[1];
        }

#pragma unroll
        for (int mt = 0; mt < 2; ++mt)
#pragma unroll
            for (int nt = 0; nt < 2; ++nt)
                acc[mt][nt] = __builtin_amdgcn_wmma_f32_16x16x32_bf16(
                    false, a[mt].v, false, bf[nt].v, (short)0, acc[mt][nt],
                    false, false);
    }

    // ---- epilogue: C layout = VGPR v -> M = v (lanes<16) / v+8 (lanes>=16) ----
    const int rbase = (lane & 16) ? 8 : 0;
#pragma unroll
    for (int mt = 0; mt < 2; ++mt) {
#pragma unroll
        for (int nt = 0; nt < 2; ++nt) {
            const int w = ni * 32 + nt * 16 + nc;
#pragma unroll
            for (int v = 0; v < 8; ++v) {
                const int co = mi2 * 32 + mt * 16 + rbase + v;
                out[(((size_t)b * COUT + co) * H_ + h) * W_ + w] =
                    acc[mt][nt][v] + bias_eff[b * COUT + co];
            }
        }
    }
}

// =====================================================================
// launcher
// =====================================================================
extern "C" void kernel_launch(void* const* d_in, const int* in_sizes, int n_in,
                              void* d_out, int out_size, void* d_ws, size_t ws_size,
                              hipStream_t stream) {
    const float* x      = (const float*)d_in[0];  // [16,128,64,64]
    const float* weight = (const float*)d_in[1];  // [512,128,3,3]
    const float* bias   = (const float*)d_in[2];  // [512]
    const float* se_w1  = (const float*)d_in[3];  // [33,128]
    const float* se_w2  = (const float*)d_in[4];  // [4,33]
    const float* se_b2  = (const float*)d_in[5];  // [4]
    float* out = (float*)d_out;

    // workspace layout
    unsigned short* wA = (unsigned short*)d_ws;                   // 4,718,592 B
    char* base = (char*)d_ws + WA_ELEMS * sizeof(unsigned short);
    float* pooled   = (float*)base;                               // 16*128
    float* attn     = pooled + B_ * CIN;                          // 16*4
    float* bias_eff = attn + B_ * KBR;                            // 16*128

    k_pool<<<B_ * CIN, 256, 0, stream>>>(x, pooled);
    k_se<<<1, 256, 0, stream>>>(pooled, se_w1, se_w2, se_b2, bias, attn, bias_eff);
    k_wcomb<<<dim3(KSTEPS, B_), 256, 0, stream>>>(weight, attn, wA);
    k_conv<<<dim3(H_, B_), 256, 0, stream>>>(x, wA, bias_eff, out);
}
